// ConcatCritic_56453050138736
// MI455X (gfx1250) — compile-verified, matrix operand order checked by reading, not run on hardware
//
#include <hip/hip_runtime.h>
#include <hip/hip_bf16.h>

#define BATCH 512
#define DIMX  128
#define HID   512
#define LDA   520   // padded LDS row stride in bf16 elements (1040B: conflict-free b128 loads)
#define RSTR  18    // reduction buffer row stride (floats): half-wave bank groups disjoint

typedef __attribute__((ext_vector_type(16))) __bf16 bf16x16;
typedef __attribute__((ext_vector_type(8)))  float  floatx8;

union Frag {
    bf16x16 f;
    uint4   q[2];
};

__device__ __forceinline__ unsigned short f2bf(float x) {
    unsigned int u = __float_as_uint(x);
    u += 0x7FFFu + ((u >> 16) & 1u);      // round-to-nearest-even
    return (unsigned short)(u >> 16);
}

// Kernel 1: hx = x @ W1[:128]  ;  hyb = y @ W1[128:] + b1   (f32, tiny: 134 MFLOP)
__global__ void precompute_kernel(const float* __restrict__ x, const float* __restrict__ y,
                                  const float* __restrict__ W1, const float* __restrict__ b1,
                                  float* __restrict__ hx, float* __restrict__ hyb) {
    const int i   = blockIdx.x;
    const int tid = threadIdx.x;
    for (int h = tid; h < HID; h += 256) {
        float sx = 0.f, sy = 0.f;
        #pragma unroll 4
        for (int d = 0; d < DIMX; ++d) {
            sx = fmaf(x[i * DIMX + d], W1[d * HID + h], sx);
            sy = fmaf(y[i * DIMX + d], W1[(DIMX + d) * HID + h], sy);
        }
        hx[i * HID + h]  = sx;
        hyb[i * HID + h] = sy + b1[h];
    }
}

// Kernel 1b: W2T_bf16[n][k] = bf16(W2[k][n])  (one-time 1MB transpose)
__global__ void w2_transpose_kernel(const float* __restrict__ W2,
                                    unsigned short* __restrict__ w2t) {
    const int e = blockIdx.x * blockDim.x + threadIdx.x;   // 0 .. HID*HID-1
    const int n = e >> 9;
    const int k = e & (HID - 1);
    w2t[n * HID + k] = f2bf(W2[k * HID + n]);
}

// Main fused kernel: 64 pairs (fixed i, j0..j0+63) per workgroup, 16 waves.
// Wave w owns all 64 M rows x N columns [w*32, w*32+32)  -> W2T read once per WG.
__global__ __launch_bounds__(512, 1) void critic_main_kernel(
    const float* __restrict__ hx, const float* __restrict__ hyb,
    const unsigned short* __restrict__ w2t,
    const float* __restrict__ b2, const float* __restrict__ W3,
    const float* __restrict__ b3, float* __restrict__ out)
{
    __shared__ __align__(16) unsigned short smA[64 * LDA];
    __shared__ float red2[64 * RSTR];

    const int tid  = threadIdx.x;
    const int lane = tid & 31;
    const int wave = tid >> 5;                 // 0..15
    const int i    = blockIdx.x >> 3;
    const int j0   = (blockIdx.x & 7) << 6;

    // zero the per-(m, lane-column) partial-sum buffer
    #pragma unroll
    for (int t = tid; t < 64 * RSTR; t += 512) red2[t] = 0.f;

    // Build A tile: h1[m][k] = relu(hx[i][k] + hy[j0+m][k] + b1[k]) in bf16.
    // tid == k (512 threads == 512 k's), loop over the 64 rows: fully coalesced.
    {
        const float hxk = hx[i * HID + tid];
        #pragma unroll 4
        for (int m = 0; m < 64; ++m) {
            float v = hxk + hyb[(j0 + m) * HID + tid];
            smA[m * LDA + tid] = f2bf(fmaxf(v, 0.f));
        }
    }
    __syncthreads();

    const int nbase = wave << 5;
    const int lhalf = (lane >> 4) << 3;        // 0 (lanes 0-15) or 8 (lanes 16-31)
    const int lrow  = lane & 15;               // M row (A) / N column (B) within tile

    floatx8 acc[4][2];
    const floatx8 vzero = {0.f, 0.f, 0.f, 0.f, 0.f, 0.f, 0.f, 0.f};
    #pragma unroll
    for (int mt = 0; mt < 4; ++mt) {
        acc[mt][0] = vzero;
        acc[mt][1] = vzero;
    }

    const unsigned short* bptr0 = w2t + (nbase + lrow) * HID;
    const unsigned short* bptr1 = w2t + (nbase + 16 + lrow) * HID;

    #pragma unroll 1
    for (int kb = 0; kb < HID; kb += 32) {
        const int koff = kb + lhalf;           // K-packing per ISA 16-bit A/B layout
        Frag a[4], bA, bB;
        #pragma unroll
        for (int mt = 0; mt < 4; ++mt) {
            const unsigned short* p = &smA[(mt * 16 + lrow) * LDA + koff];
            a[mt].q[0] = *(const uint4*)(p);
            a[mt].q[1] = *(const uint4*)(p + 16);
        }
        bA.q[0] = *(const uint4*)(bptr0 + koff);
        bA.q[1] = *(const uint4*)(bptr0 + koff + 16);
        bB.q[0] = *(const uint4*)(bptr1 + koff);
        bB.q[1] = *(const uint4*)(bptr1 + koff + 16);

        #pragma unroll
        for (int mt = 0; mt < 4; ++mt) {
            acc[mt][0] = __builtin_amdgcn_wmma_f32_16x16x32_bf16(
                false, a[mt].f, false, bA.f, (short)0, acc[mt][0], false, false);
            acc[mt][1] = __builtin_amdgcn_wmma_f32_16x16x32_bf16(
                false, a[mt].f, false, bB.f, (short)0, acc[mt][1], false, false);
        }
    }

    // Epilogue: score[m] += sum_n relu(h2[m][n] + b2[n]) * W3[n]
    // Each wave-wide ds_add_f32 hits 32 distinct, bank-disjoint LDS addresses:
    // address = (mt*16 + r + lhalf) * RSTR + lrow.  Cross-wave sums via atomics.
    const float b2v0 = b2[nbase + lrow];
    const float b2v1 = b2[nbase + 16 + lrow];
    const float w3v0 = W3[nbase + lrow];
    const float w3v1 = W3[nbase + 16 + lrow];

    #pragma unroll
    for (int mt = 0; mt < 4; ++mt) {
        #pragma unroll
        for (int r = 0; r < 8; ++r) {
            float s = fmaxf(acc[mt][0][r] + b2v0, 0.f) * w3v0
                    + fmaxf(acc[mt][1][r] + b2v1, 0.f) * w3v1;
            atomicAdd(&red2[(mt * 16 + r + lhalf) * RSTR + lrow], s);
        }
    }
    __syncthreads();

    if (tid < 64) {
        float s = b3[0];
        #pragma unroll
        for (int c = 0; c < 16; ++c) s += red2[tid * RSTR + c];
        out[i * BATCH + j0 + tid] = s;
    }
}

extern "C" void kernel_launch(void* const* d_in, const int* in_sizes, int n_in,
                              void* d_out, int out_size, void* d_ws, size_t ws_size,
                              hipStream_t stream) {
    (void)in_sizes; (void)n_in; (void)out_size; (void)ws_size;
    const float* x  = (const float*)d_in[0];
    const float* y  = (const float*)d_in[1];
    const float* W1 = (const float*)d_in[2];
    const float* b1 = (const float*)d_in[3];
    const float* W2 = (const float*)d_in[4];
    const float* b2 = (const float*)d_in[5];
    const float* W3 = (const float*)d_in[6];
    const float* b3 = (const float*)d_in[7];
    float* out = (float*)d_out;

    float* hx  = (float*)d_ws;                         // 512*512 f32
    float* hyb = hx + BATCH * HID;                     // 512*512 f32
    unsigned short* w2t = (unsigned short*)(hyb + BATCH * HID);  // 512*512 bf16

    precompute_kernel<<<BATCH, 256, 0, stream>>>(x, y, W1, b1, hx, hyb);
    w2_transpose_kernel<<<(HID * HID) / 256, 256, 0, stream>>>(W2, w2t);
    critic_main_kernel<<<(BATCH * BATCH) / 64, 512, 0, stream>>>(hx, hyb, w2t, b2, W3, b3, out);
}
